// LSTEmbedding_3453153706135
// MI455X (gfx1250) — matrix operand order repaired and checked by reading, not compiled, not run on hardware
//
#include <hip/hip_runtime.h>
#include <stdint.h>

// Problem constants: N,S,D,E,L = 16,2048,256,512,4
#define N_   16
#define S_   2048
#define D_   256
#define E_   512
#define G_   2048            // 4*E
#define L_   4
#define NWG  64              // 4 layer-groups x 16 WGs
#define UNITS_PER_WG 32
#define TPB  512             // 16 wave32: 8 N-tiles x 2 K-halves

typedef __bf16 bf16;
typedef __attribute__((ext_vector_type(16))) __bf16 v16bf;
typedef __attribute__((ext_vector_type(8)))  float  v8f;

// ---- workspace layout (bytes) ----
// [0..7]      : int bar_cnt, bar_gen
// [64..127]   : int nsegs[16]
// [512..]     : int slot[16][2048]                 (131072 B)
// [131584..]  : bf16 hbuf[2][4][8192]              (131072 B)
//   h stored pre-swizzled in WMMA A-operand layout: [16 et][32 ln][16 j]
#define WS_NSEGS_OFF_I32   16
#define WS_SLOT_OFF_I32    128
#define WS_HBUF_OFF_BYTES  131584
#define HBUF_U32_WORDS     32768
#define HSWZ_ELEMS         8192          // bf16 per (parity, layer)

// LDS: sA [32 kt][32 ln][16] bf16 (32 KB) + sG [128 gates][16 batch] f32 (8 KB)
#define SA_BYTES   (32*32*16*2)
#define SG_BYTES   (128*16*4)
#define LDS_TOTAL  (SA_BYTES + SG_BYTES)   // 40960 B

// --------------------------------------------------------------------------
__global__ void lstm_prep_kernel(const uint8_t* __restrict__ mask, int* __restrict__ ws_i) {
  const int n = threadIdx.x;
  if (n == 0) { ws_i[0] = 0; ws_i[1] = 0; }
  if (n < N_) {
    int* nsegs = ws_i + WS_NSEGS_OFF_I32;
    int* srow  = ws_i + WS_SLOT_OFF_I32 + n * S_;
    const uint8_t* mrow = mask + (size_t)n * S_;
    for (int t = 0; t < S_; ++t) srow[t] = -1;
    int i = 0, cnt = 0;
    for (int t = 0; t < S_; ++t) {
      if (mrow[t]) { if (i != t) srow[t - 1] = cnt++; i = t + 1; }
    }
    if (i != S_) srow[S_ - 1] = cnt++;
    nsegs[n] = cnt;
  }
}

// --------------------------------------------------------------------------
__global__ void lstm_init_kernel(float* __restrict__ out, int Sp,
                                 const int* __restrict__ nsegs,
                                 uint32_t* __restrict__ hbuf_u32) {
  const long total_ret  = (long)N_ * Sp * E_;
  const long total_mask = (long)N_ * Sp;
  const long total      = total_ret + total_mask + HBUF_U32_WORDS;
  for (long idx = (long)blockIdx.x * blockDim.x + threadIdx.x; idx < total;
       idx += (long)gridDim.x * blockDim.x) {
    if (idx < total_ret) {
      out[idx] = 0.0f;
    } else if (idx < total_ret + total_mask) {
      long m = idx - total_ret;
      int n = (int)(m / Sp), s = (int)(m % Sp);
      out[idx] = (s < nsegs[n]) ? 0.0f : 1.0f;
    } else {
      hbuf_u32[idx - total_ret - total_mask] = 0u;
    }
  }
}

// --------------------------------------------------------------------------
// Async 16KB contiguous copy global -> LDS (ASYNCcnt path, GVS addressing).
// 512 threads x 2 iters x 16B = 16384 B.
__device__ __forceinline__ void async_copy_16k(uint32_t ldsOff, const void* src, int tid) {
#pragma unroll
  for (int it = 0; it < 2; ++it) {
    const uint32_t off = (uint32_t)(it * 8192 + tid * 16);
    const uint32_t la  = ldsOff + off;
    asm volatile("global_load_async_to_lds_b128 %0, %1, %2"
                 :: "v"(la), "v"(off), "s"(src) : "memory");
  }
}

// --------------------------------------------------------------------------
// Software-pipelined K-loop: 3-slot rotating A buffer so loads for kt+1/kt+2
// are in flight while wmma(kt) executes (s_wait_dscnt <= 2 instead of 0).
template <int NSTEPS>
__device__ __forceinline__ v8f gemm_pipe(const char* aBw, const v16bf (&breg)[16]) {
  v8f acc = {};
  v16bf ab[3];
  ab[0] = *(const v16bf*)(aBw);
  if (NSTEPS > 1) ab[1] = *(const v16bf*)(aBw + 1024);
#pragma unroll
  for (int kt = 0; kt < NSTEPS; ++kt) {
    if (kt + 2 < NSTEPS)
      ab[(kt + 2) % 3] = *(const v16bf*)(aBw + (size_t)(kt + 2) * 1024);
    acc = __builtin_amdgcn_wmma_f32_16x16x32_bf16(
        false, ab[kt % 3], false, breg[kt], (short)0, acc, false, false);
  }
  return acc;
}

// --------------------------------------------------------------------------
__global__ __launch_bounds__(TPB, 1)
void lstm_persistent_kernel(const float* __restrict__ x,
                            const uint8_t* __restrict__ mask,
                            const float* __restrict__ Wih0,
                            const float* __restrict__ Wihr,
                            const float* __restrict__ Whh,
                            const float* __restrict__ bih,
                            const float* __restrict__ bhh,
                            float* __restrict__ out, int Sp,
                            const int* __restrict__ slot,
                            bf16* __restrict__ hbuf,
                            int* __restrict__ bar) {
  extern __shared__ char smem[];
  bf16*  sA = (bf16*)smem;                    // [32 kt][32 ln][16]
  float* sG = (float*)(smem + SA_BYTES);      // [128 gate][16 batch]

  const int tid   = threadIdx.x;
  const int wave  = tid >> 5;                 // 0..15
  const int lane  = tid & 31;
  const int wt    = wave & 7;                 // N-tile (16 gates)
  const int ksel  = wave >> 3;                // 0: kt 0..15 (input), 1: kt 16..31 (h)
  const int layer = blockIdx.x >> 4;
  const int u0    = (blockIdx.x & 15) * UNITS_PER_WG;

  // ---- one-time: this wave's half-K weight slice -> 16 x v16bf registers ----
  // B 32x16 tile: lane holds column n = lane&15; within-tile K = (lane>>4)*16+j
  // => each lane's 16 K-values are CONTIGUOUS in the fp32 weight row.
  v16bf breg[16];
  {
    const int r_local = wt * 16 + (lane & 15);                     // 0..127
    const int growB   = (r_local >> 5) * E_ + u0 + (r_local & 31); // global gate row
    const int khalf   = (lane >> 4) * 16;
    const float* WihL = (layer == 0) ? (Wih0 + (size_t)growB * D_)
                                     : (Wihr + ((size_t)(layer - 1) * G_ + growB) * E_);
    const float* WhhL = Whh + ((size_t)layer * G_ + growB) * E_;
#pragma unroll
    for (int kt = 0; kt < 16; ++kt) {
      const int kg = (ksel * 16 + kt) * 32 + khalf;  // global K start (per lane half)
      v16bf b = {};
      if (ksel == 0) {
        if (layer == 0) {
          if (kt < 8) {                                // kg < 256
            const float* s = WihL + kg;
#pragma unroll
            for (int j = 0; j < 16; ++j) b[j] = (bf16)s[j];
          }                                            // kt 8..15: zero pad
        } else {                                       // kg < 512
          const float* s = WihL + kg;
#pragma unroll
          for (int j = 0; j < 16; ++j) b[j] = (bf16)s[j];
        }
      } else {                                         // h-recurrence half
        const float* s = WhhL + (kg - 512);
#pragma unroll
        for (int j = 0; j < 16; ++j) b[j] = (bf16)s[j];
      }
      breg[kt] = b;
    }
  }

  // ---- per-thread fixed (unit, batch) state for the pointwise stage ----
  const int u_t = tid >> 4;           // 0..31 hidden unit (local)
  const int m_t = tid & 15;           // batch row
  const int eg  = u0 + u_t;           // global hidden unit
  const int et = eg >> 5, cA = eg & 31;
  const int ccA = cA >> 4, halfA = (cA >> 3) & 1, jA = cA & 7;
  // bias for the sG row this thread re-initializes each tick (row = tid/4)
  const int r_t   = tid >> 2;
  const int growR = (r_t >> 5) * E_ + u0 + (r_t & 31);
  const float biasR = bih[layer * G_ + growR] + bhh[layer * G_ + growR];
  float cS = 0.0f;

  const uint32_t saOff = (uint32_t)(uintptr_t)sA;   // LDS byte offset (addr[31:0])
  const char* aBw = (const char*)sA + ksel * 16384 + lane * 32;

  for (int k = 0; k < S_ + L_ - 1; ++k) {
    const int t = k - layer;
    const bool active = (t >= 0) && (t < S_);
    if (active) {
      const int parR = (k + 1) & 1;  // parity written at tick k-1
      const bf16* hOwnSwz = hbuf + ((size_t)parR * L_ + layer) * HSWZ_ELEMS;

      // ---- stage A: contiguous async copies of pre-swizzled h ----
      if (layer != 0) {
        const bf16* hPrevSwz = hbuf + ((size_t)parR * L_ + (layer - 1)) * HSWZ_ELEMS;
        async_copy_16k(saOff, hPrevSwz, tid);            // kt 0..15
      }
      async_copy_16k(saOff + 16384, hOwnSwz, tid);       // kt 16..31
      if (layer == 0) {
        // x (fp32) -> bf16 swizzle into kt 0..7 (k < 256); one chunk/thread
        const int kt = tid >> 6, rem = tid & 63;
        const int ln = rem >> 1, cc = rem & 1;
        const int m = ln & 15;
        const int kbase = kt * 32 + ((ln >> 4) << 4) + cc * 8;
        const float* src = x + ((size_t)m * S_ + t) * D_ + kbase;
        bf16* dst = sA + ((size_t)kt * 32 + ln) * 16 + cc * 8;
#pragma unroll
        for (int j = 0; j < 8; ++j) dst[j] = (bf16)src[j];
      }
      // re-init accumulator LDS to the gate bias (4 consecutive f32 / thread)
      {
        float* gi = sG + tid * 4;
        gi[0] = biasR; gi[1] = biasR; gi[2] = biasR; gi[3] = biasR;
      }
      asm volatile("s_wait_asynccnt 0x0" ::: "memory");
      __syncthreads();

      // ---- GEMM: software-pipelined half-K per wave; B register-resident ----
      v8f acc;
      if (layer == 0 && ksel == 0) acc = gemm_pipe<8>(aBw, breg);
      else                         acc = gemm_pipe<16>(aBw, breg);
      // reduce both K-halves: ds_add_f32 into sG (pre-set to bias)
      // C layout: lane -> gate n = lane&15; VGPR r -> m = r + 8*(lane>>4)
      {
        float* g = sG + ((size_t)wt * 16 + (lane & 15)) * 16 + (lane >> 4) * 8;
#pragma unroll
        for (int r = 0; r < 8; ++r) atomicAdd(&g[r], acc[r]);
      }
      __syncthreads();

      // ---- fused LSTM pointwise: one (unit,batch) state per thread ----
      const int par = k & 1;
      bf16* hOut = hbuf + ((size_t)par * L_ + layer) * HSWZ_ELEMS;
      {
        const float gi = sG[(u_t     ) * 16 + m_t];
        const float gf = sG[(32 + u_t) * 16 + m_t];
        const float gg = sG[(64 + u_t) * 16 + m_t];
        const float go = sG[(96 + u_t) * 16 + m_t];
        const float si = 1.0f / (1.0f + __expf(-gi));
        const float sf = 1.0f / (1.0f + __expf(-gf));
        const float so = 1.0f / (1.0f + __expf(-go));
        float cN = sf * cS + si * tanhf(gg);
        float h  = so * tanhf(cN);
        if (layer == L_ - 1) {                       // emit y BEFORE reset
          const int sl = slot[m_t * S_ + t];
          if (sl >= 0) out[((size_t)m_t * Sp + sl) * E_ + eg] = h;
        }
        if (mask[m_t * S_ + t]) { h = 0.0f; cN = 0.0f; }
        cS = cN;
        hOut[((size_t)(et * 32) + m_t + 16 * ccA) * 16 + halfA * 8 + jA] = (bf16)h;
      }
      if (layer == 0 && t + 1 < S_) {                // prefetch next x slab
        const float* pf = x + ((size_t)(tid & 15) * S_ + (t + 1)) * D_
                            + (((tid >> 4) & 15) << 4);
        __builtin_prefetch(pf, 0, 2);
      }
    }

    // ---- grid barrier (sense-reversing, agent scope) ----
    __threadfence();
    __syncthreads();
    if (tid == 0) {
      const int gen = __hip_atomic_load(bar + 1, __ATOMIC_RELAXED, __HIP_MEMORY_SCOPE_AGENT);
      const int arr = __hip_atomic_fetch_add(bar, 1, __ATOMIC_ACQ_REL, __HIP_MEMORY_SCOPE_AGENT);
      if (arr == NWG - 1) {
        __hip_atomic_store(bar, 0, __ATOMIC_RELAXED, __HIP_MEMORY_SCOPE_AGENT);
        __hip_atomic_fetch_add(bar + 1, 1, __ATOMIC_RELEASE, __HIP_MEMORY_SCOPE_AGENT);
      } else {
        while (__hip_atomic_load(bar + 1, __ATOMIC_ACQUIRE, __HIP_MEMORY_SCOPE_AGENT) == gen)
          __builtin_amdgcn_s_sleep(1);
      }
    }
    __syncthreads();
  }
}

// --------------------------------------------------------------------------
extern "C" void kernel_launch(void* const* d_in, const int* in_sizes, int n_in,
                              void* d_out, int out_size, void* d_ws, size_t ws_size,
                              hipStream_t stream) {
  (void)in_sizes; (void)n_in; (void)ws_size;
  const float*   x    = (const float*)d_in[0];
  const uint8_t* mask = (const uint8_t*)d_in[1];
  const float*   Wih0 = (const float*)d_in[2];
  const float*   Wihr = (const float*)d_in[3];
  const float*   Whh  = (const float*)d_in[4];
  const float*   bih  = (const float*)d_in[5];
  const float*   bhh  = (const float*)d_in[6];
  float* out = (float*)d_out;
  const int Sp = out_size / (N_ * (E_ + 1));

  int*  ws_i  = (int*)d_ws;
  int*  bar   = ws_i;
  int*  nsegs = ws_i + WS_NSEGS_OFF_I32;
  int*  slot  = ws_i + WS_SLOT_OFF_I32;
  bf16* hbuf  = (bf16*)((char*)d_ws + WS_HBUF_OFF_BYTES);

  lstm_prep_kernel<<<1, 32, 0, stream>>>(mask, ws_i);
  lstm_init_kernel<<<256, 256, 0, stream>>>(out, Sp, nsegs, (uint32_t*)hbuf);
  lstm_persistent_kernel<<<NWG, TPB, LDS_TOTAL, stream>>>(
      x, mask, Wih0, Wihr, Whh, bih, bhh, out, Sp, slot, hbuf, bar);
}
